// MsaHmmLayer_83966610637188
// MI455X (gfx1250) — matrix operand order repaired and checked by reading, not compiled, not run on hardware
//
#include <hip/hip_runtime.h>
#include <hip/hip_bf16.h>
#include <math.h>

typedef __attribute__((ext_vector_type(16))) __bf16 v16bf;
typedef __attribute__((ext_vector_type(8)))  __bf16 v8bf;
typedef __attribute__((ext_vector_type(8)))  float  v8f;

#define MM 4
#define BB 64
#define LL 512
#define QQ 512
#define SS 26
#define SP 32   // S padded to WMMA K granularity

// Build a v16bf from two 16-byte chunks (matches the 16-bit A/B operand lane
// layouts, which split into two 8-element K groups).
__device__ __forceinline__ v16bf ld16(const __bf16* p0, const __bf16* p1) {
    v8bf lo = *(const v8bf*)p0;
    v8bf hi = *(const v8bf*)p1;
    return __builtin_shufflevector(lo, hi, 0,1,2,3,4,5,6,7,8,9,10,11,12,13,14,15);
}

// ---------------------------------------------------------------------------
// Prep: convert inputs [M,B,L,S] f32 -> [M,B,L,SP] bf16 (zero-padded).
// Pure bandwidth kernel; each thread produces one 16B (v8bf) chunk.
// ---------------------------------------------------------------------------
__global__ void pack_obs(const float* __restrict__ inputs,
                         __bf16* __restrict__ obs) {
    // M*B*L rows, 4 chunks of 8 per row.
    size_t idx = (size_t)blockIdx.x * blockDim.x + threadIdx.x;
    size_t nChunks = (size_t)MM * BB * LL * (SP / 8);
    if (idx >= nChunks) return;
    size_t row = idx >> 2;
    int c = (int)(idx & 3);
    const float* src = inputs + row * SS + c * 8;
    int valid = SS - c * 8;           // how many of the 8 are in-range
    v8bf v;
#pragma unroll
    for (int i = 0; i < 8; ++i)
        v[i] = (__bf16)((i < valid) ? src[i] : 0.f);
    *(v8bf*)(obs + row * SP + c * 8) = v;
}

// ---------------------------------------------------------------------------
// Prep: row-softmax of A_logits [M,Q,Q], packed into WMMA B-operand tiles.
// Ap[(((m*16 + kt)*32 + nt)*32 + lane)*16 + e], where for element (k,n):
// kt=k>>5, e=k&15, lh=(k>>4)&1, lane=(n&15)+16*lh, nt=n>>4.
// ---------------------------------------------------------------------------
__global__ void softmax_pack_A(const float* __restrict__ A_logits,
                               __bf16* __restrict__ Ap) {
    int rowid = blockIdx.x;                 // m*Q + k
    int m = rowid / QQ;
    int k = rowid % QQ;
    const float* src = A_logits + (size_t)rowid * QQ;
    __shared__ float red[256];
    int tid = threadIdx.x;
    float x0 = src[tid], x1 = src[tid + 256];
    red[tid] = fmaxf(x0, x1); __syncthreads();
    for (int off = 128; off > 0; off >>= 1) {
        if (tid < off) red[tid] = fmaxf(red[tid], red[tid + off]);
        __syncthreads();
    }
    float mx = red[0]; __syncthreads();
    float e0 = __expf(x0 - mx), e1 = __expf(x1 - mx);
    red[tid] = e0 + e1; __syncthreads();
    for (int off = 128; off > 0; off >>= 1) {
        if (tid < off) red[tid] += red[tid + off];
        __syncthreads();
    }
    float inv = 1.0f / red[0];
    int kt = k >> 5, e = k & 15, lh = (k >> 4) & 1;
    for (int n = tid; n < QQ; n += 256) {
        float v = (n == tid ? e0 : e1) * inv;
        int nt = n >> 4;
        int lanept = (n & 15) + 16 * lh;
        size_t off2 = ((((size_t)m * 16 + kt) * 32 + nt) * 32 + lanept) * 16 + e;
        Ap[off2] = (__bf16)v;
    }
}

// ---------------------------------------------------------------------------
// Prep: softmax of em_logits [M,Q,S] over S, packed as B-operand (k = s).
// ---------------------------------------------------------------------------
__global__ void softmax_pack_Bem(const float* __restrict__ em_logits,
                                 __bf16* __restrict__ Bp) {
    int rowid = blockIdx.x;                 // m*Q + q
    int m = rowid / QQ, q = rowid % QQ;
    int s = threadIdx.x;                    // 0..31 (one wave)
    float x = (s < SS) ? em_logits[(size_t)rowid * SS + s] : -1e30f;
    float mx = x;
    for (int off = 16; off > 0; off >>= 1) mx = fmaxf(mx, __shfl_xor(mx, off));
    float e = __expf(x - mx);
    float sum = e;
    for (int off = 16; off > 0; off >>= 1) sum += __shfl_xor(sum, off);
    float v = e / sum;
    int nt = q >> 4;
    int lanept = (q & 15) + 16 * (s >> 4);
    int ee = s & 15;
    Bp[(((size_t)m * 32 + nt) * 32 + lanept) * 16 + ee] = (__bf16)v;
}

// ---------------------------------------------------------------------------
// Prep: softmax of init_logits [M,Q] -> f32 pi.
// ---------------------------------------------------------------------------
__global__ void softmax_pi(const float* __restrict__ init_logits,
                           float* __restrict__ pi) {
    int m = blockIdx.x;
    int tid = threadIdx.x;                  // 0..511
    __shared__ float red[512];
    float x = init_logits[(size_t)m * QQ + tid];
    red[tid] = x; __syncthreads();
    for (int off = 256; off > 0; off >>= 1) {
        if (tid < off) red[tid] = fmaxf(red[tid], red[tid + off]);
        __syncthreads();
    }
    float mx = red[0]; __syncthreads();
    float e = __expf(x - mx);
    red[tid] = e; __syncthreads();
    for (int off = 256; off > 0; off >>= 1) {
        if (tid < off) red[tid] += red[tid + off];
        __syncthreads();
    }
    pi[(size_t)m * QQ + tid] = e / red[0];
}

// ---------------------------------------------------------------------------
// Scaled forward recurrence. Grid = 16 WGs (4 per model, each a 16-row batch
// tile; batch rows are independent chains -> zero inter-WG traffic).
// 8 waves/WG; each wave owns 4 n-tiles (64 states). Per step: emission via
// 4 WMMAs (K=32 pad of S, A-operand straight from global bf16 obs), and the
// transition GEMM via 64 WMMAs (16 k-tiles x 4 n-tiles) streaming packed A
// from L2 (2 MB total, fully L2-resident). 3 barriers/step.
// ---------------------------------------------------------------------------
__global__ void __launch_bounds__(256, 1)
hmm_forward(const __bf16* __restrict__ obs, const __bf16* __restrict__ Ap,
            const __bf16* __restrict__ Bp, const float* __restrict__ pi,
            float* __restrict__ out) {
    __shared__ __align__(32) __bf16 s_bem[32 * 32 * 16];   // 32 KB, B-op layout
    __shared__ __align__(32) __bf16 s_alpha[16 * QQ];      // 16 KB, row-major
    __shared__ float s_rowsum[16];
    __shared__ float s_inv[16];
    __shared__ float s_ll[16];

    const int wg   = blockIdx.x;
    const int m    = wg >> 2;
    const int b0   = (wg & 3) * 16;
    const int tid  = threadIdx.x;
    const int lane = tid & 31;
    const int wv   = tid >> 5;
    const int lcol = lane & 15;     // row for A-operand, col for B/C operands
    const int lh   = lane >> 4;
    const int nt0  = wv * 4;

    // Load this model's emission panel into LDS (once).
    {
        const v8bf* src = (const v8bf*)(Bp + (size_t)m * 32 * 32 * 16);
        v8bf* dst = (v8bf*)s_bem;
        for (int i = tid; i < (32 * 32 * 16) / 8; i += 256) dst[i] = src[i];
    }
    if (tid < 16) { s_rowsum[tid] = 0.f; s_ll[tid] = 0.f; }
    __syncthreads();

    // Emission B-operands are time-invariant: keep in registers.
    v16bf ebv[4];
#pragma unroll
    for (int j = 0; j < 4; ++j) {
        const __bf16* p = &s_bem[((nt0 + j) * 32 + lane) * 16];
        ebv[j] = ld16(p, p + 8);
    }
    // pi values for t==0 (depend only on the column).
    float piv[4];
#pragma unroll
    for (int j = 0; j < 4; ++j)
        piv[j] = pi[(size_t)m * QQ + (nt0 + j) * 16 + lcol];

    const __bf16* ApBase = Ap + (size_t)m * 16 * 32 * 32 * 16;
    // Per-lane emission A-operand pointer: row b0+lcol, advances SP per step.
    const __bf16* obsPtr =
        obs + ((size_t)m * BB + (b0 + lcol)) * LL * SP + lh * 16;

    for (int t = 0; t < LL; ++t) {
        // Issue emission A-operand loads early; they ride under the k-loop.
        v16bf ea = ld16(obsPtr, obsPtr + 8);
        obsPtr += SP;

        // Transition: r = alpha @ A  (A streamed from L2 in B-op layout).
        v8f acc[4];
        if (t == 0) {
#pragma unroll
            for (int j = 0; j < 4; ++j)
#pragma unroll
                for (int v = 0; v < 8; ++v) acc[j][v] = piv[j];
        } else {
#pragma unroll
            for (int j = 0; j < 4; ++j) {
                v8f z = {0.f, 0.f, 0.f, 0.f, 0.f, 0.f, 0.f, 0.f};
                acc[j] = z;
            }
            for (int kt = 0; kt < 16; ++kt) {
                const __bf16* ar = &s_alpha[lcol * QQ + kt * 32];
                v16bf av = ld16(ar + lh * 16, ar + lh * 16 + 8);
                const __bf16* bp =
                    ApBase + (((size_t)kt * 32 + nt0) * 32 + lane) * 16;
#pragma unroll
                for (int j = 0; j < 4; ++j) {
                    v16bf bv = ld16(bp, bp + 8);
                    acc[j] = __builtin_amdgcn_wmma_f32_16x16x32_bf16(
                        false, av, false, bv, (short)0, acc[j], false, false);
                    bp += 32 * 16;   // next n-tile
                }
            }
        }

        // Emission tile: Et = obs_t @ Bem^T (one K=32 WMMA per n-tile).
        v8f et[4];
#pragma unroll
        for (int j = 0; j < 4; ++j) {
            v8f z = {0.f, 0.f, 0.f, 0.f, 0.f, 0.f, 0.f, 0.f};
            et[j] = __builtin_amdgcn_wmma_f32_16x16x32_bf16(
                false, ea, false, ebv[j], (short)0, z, false, false);
        }

        // a = r * e_t ; per-row partial sums (C layout: row = v + 8*lh).
        float aval[4][8];
        float part[8];
#pragma unroll
        for (int v = 0; v < 8; ++v) part[v] = 0.f;
#pragma unroll
        for (int j = 0; j < 4; ++j)
#pragma unroll
            for (int v = 0; v < 8; ++v) {
                float x = acc[j][v] * et[j][v];
                aval[j][v] = x;
                part[v] += x;
            }
        // Reduce across the 16 lanes of each half (same rows, distinct cols).
#pragma unroll
        for (int v = 0; v < 8; ++v) {
#pragma unroll
            for (int off = 1; off < 16; off <<= 1)
                part[v] += __shfl_xor(part[v], off);
        }
        if (lcol == 0) {
#pragma unroll
            for (int v = 0; v < 8; ++v)
                atomicAdd(&s_rowsum[v + lh * 8], part[v]);
        }
        __syncthreads();

        if (tid < 16) {
            float s = s_rowsum[tid];
            s_ll[tid] += logf(s);
            s_inv[tid] = 1.0f / s;
            s_rowsum[tid] = 0.f;    // ready for next step
        }
        __syncthreads();

        // Rescale and store alpha as bf16 for the next step's A-operand.
#pragma unroll
        for (int j = 0; j < 4; ++j) {
            int col = (nt0 + j) * 16 + lcol;
#pragma unroll
            for (int v = 0; v < 8; ++v) {
                int row = v + lh * 8;
                s_alpha[row * QQ + col] = (__bf16)(aval[j][v] * s_inv[row]);
            }
        }
        __syncthreads();
    }

    if (tid < 16) out[(size_t)m * BB + b0 + tid] = s_ll[tid];
}

// ---------------------------------------------------------------------------
extern "C" void kernel_launch(void* const* d_in, const int* in_sizes, int n_in,
                              void* d_out, int out_size, void* d_ws,
                              size_t ws_size, hipStream_t stream) {
    const float* inputs      = (const float*)d_in[0];   // [M,B,L,S]
    const float* A_logits    = (const float*)d_in[1];   // [M,Q,Q]
    const float* init_logits = (const float*)d_in[2];   // [M,Q]
    const float* em_logits   = (const float*)d_in[3];   // [M,Q,S]
    float* out = (float*)d_out;                         // [M,B]

    char* ws = (char*)d_ws;
    const size_t apBytes  = (size_t)MM * QQ * QQ * 2;          // 2 MB
    const size_t bemBytes = (size_t)MM * 32 * 32 * 16 * 2;     // 128 KB
    const size_t piBytes  = (size_t)MM * QQ * 4;               // 8 KB
    __bf16* Ap  = (__bf16*)ws;
    __bf16* Bp  = (__bf16*)(ws + apBytes);
    float*  piw = (float*)(ws + apBytes + bemBytes);
    __bf16* obs = (__bf16*)(ws + apBytes + bemBytes + piBytes); // 8 MB

    const size_t nChunks = (size_t)MM * BB * LL * (SP / 8);
    pack_obs        <<<(unsigned)((nChunks + 255) / 256), 256, 0, stream>>>(inputs, obs);
    softmax_pack_A  <<<MM * QQ, 256, 0, stream>>>(A_logits, Ap);
    softmax_pack_Bem<<<MM * QQ,  32, 0, stream>>>(em_logits, Bp);
    softmax_pi      <<<MM,      512, 0, stream>>>(init_logits, piw);
    hmm_forward     <<<MM * 4,  256, 0, stream>>>(obs, Ap, Bp, piw, out);
}
